// Attention_42056319762906
// MI455X (gfx1250) — compile-verified, matrix operand order checked by reading, not run on hardware
//
#include <hip/hip_runtime.h>
#include <hip/hip_bf16.h>

typedef __attribute__((ext_vector_type(16))) __bf16 v16bf;
typedef __attribute__((ext_vector_type(8)))  __bf16 v8bf;
typedef __attribute__((ext_vector_type(8)))  float  v8f;

#define DIM   4096
#define HEADS 32
#define HD    128
#define SEQ   2048
#define BSZ   2
#define NTOK  (BSZ*SEQ)   // 4096 tokens

union V16U { v16bf v; v8bf h[2]; __bf16 e[16]; };

__device__ __forceinline__ v16bf make_frag(const __bf16* lo, const __bf16* hi) {
    V16U u;
    u.h[0] = *(const v8bf*)lo;
    u.h[1] = *(const v8bf*)hi;
    return u.v;
}

// CDNA5 async memory->LDS copy (ASYNCcnt-tracked, no VGPR round trip).
// The instruction's immediate offset is added to BOTH the global and LDS
// addresses, so one address pair covers a 32B chunk with two ops.
__device__ __forceinline__ void async_copy32(unsigned ldsaddr, const void* gaddr) {
    asm volatile("global_load_async_to_lds_b128 %0, %1, off"
                 :: "v"(ldsaddr), "v"((unsigned long long)(uintptr_t)gaddr)
                 : "memory");
    asm volatile("global_load_async_to_lds_b128 %0, %1, off offset:16"
                 :: "v"(ldsaddr), "v"((unsigned long long)(uintptr_t)gaddr)
                 : "memory");
}
__device__ __forceinline__ void wait_async0() {
    asm volatile("s_wait_asynccnt 0x0" ::: "memory");
}
__device__ __forceinline__ unsigned lds_off(const void* p) {
    return (unsigned)(uintptr_t)p;   // generic LDS ptr: addr[31:0] = LDS byte offset
}

// ---------------------------------------------------------------------------
// f32 -> bf16 streaming convert (8 elements / thread, sizes are multiples of 8)
// ---------------------------------------------------------------------------
__global__ __launch_bounds__(256) void cvt_f32_bf16(const void* srcv, void* dstv) {
    const float* src = (const float*)srcv;
    __bf16*      dst = (__bf16*)dstv;
    size_t i = ((size_t)blockIdx.x * 256 + threadIdx.x) * 8;
    float4 a = *(const float4*)(src + i);
    float4 b = *(const float4*)(src + i + 4);
    __bf16 tmp[8] = { (__bf16)a.x,(__bf16)a.y,(__bf16)a.z,(__bf16)a.w,
                      (__bf16)b.x,(__bf16)b.y,(__bf16)b.z,(__bf16)b.w };
    *(v8bf*)(dst + i) = *(const v8bf*)tmp;
}

// ---------------------------------------------------------------------------
// C[m][n] = sum_k A[m][k] * W[n][k]   (A rowmajor, W rowmajor; N=K=4096)
// 128x128 block, 8 waves = 2(M) x 4(N), each wave 4x2 16x16 WMMA tiles.
// Double-buffered LDS, async global->LDS staging, one barrier per k-step.
// ---------------------------------------------------------------------------
template<bool OUT_BF16>
__global__ __launch_bounds__(256) void gemm_nt(const void* Av, const void* Wv, void* Cv) {
    const __bf16* A = (const __bf16*)Av;
    const __bf16* W = (const __bf16*)Wv;
    __shared__ __bf16 Alds[2][128 * 40];   // 32 K-cols + 8 pad
    __shared__ __bf16 Blds[2][128 * 40];

    const int tid  = threadIdx.x;
    const int lane = tid & 31;
    const int wave = tid >> 5;
    const int half = lane >> 4;
    const int cl   = lane & 15;
    const int wm   = wave >> 2;     // 0..1
    const int wn   = wave & 3;      // 0..3
    const int m0   = blockIdx.y * 128;
    const int n0   = blockIdx.x * 128;

    const int lr = tid >> 1;            // 0..127 staging row
    const int lc = (tid & 1) * 16;      // 0 / 16

    auto stage = [&](int k0, int buf) {
        async_copy32(lds_off(&Alds[buf][lr * 40 + lc]),
                     &A[(size_t)(m0 + lr) * DIM + k0 + lc]);
        async_copy32(lds_off(&Blds[buf][lr * 40 + lc]),
                     &W[(size_t)(n0 + lr) * DIM + k0 + lc]);
    };

    v8f acc[4][2] = {};
    stage(0, 0);

    const int NK = DIM / 32;
    for (int kt = 0; kt < NK; ++kt) {
        const int buf = kt & 1;
        wait_async0();          // my async writes (buf) have landed
        __syncthreads();        // everyone's landed; prev readers done
        if (kt + 1 < NK) stage((kt + 1) * 32, buf ^ 1);

        v16bf af[4], bfr[2];
#pragma unroll
        for (int i = 0; i < 4; i++) {
            int r = wm * 64 + i * 16 + cl;
            af[i] = make_frag(&Alds[buf][r * 40 + half * 8],
                              &Alds[buf][r * 40 + half * 8 + 16]);
        }
#pragma unroll
        for (int j = 0; j < 2; j++) {
            int r = wn * 32 + j * 16 + cl;
            bfr[j] = make_frag(&Blds[buf][r * 40 + half * 8],
                               &Blds[buf][r * 40 + half * 8 + 16]);
        }
#pragma unroll
        for (int i = 0; i < 4; i++)
#pragma unroll
            for (int j = 0; j < 2; j++)
                acc[i][j] = __builtin_amdgcn_wmma_f32_16x16x32_bf16(
                    false, af[i], false, bfr[j], (short)0, acc[i][j], false, false);
    }

#pragma unroll
    for (int i = 0; i < 4; i++)
#pragma unroll
        for (int j = 0; j < 2; j++)
#pragma unroll
            for (int v = 0; v < 8; v++) {
                int row = m0 + wm * 64 + i * 16 + v + 8 * half;
                int col = n0 + wn * 32 + j * 16 + cl;
                if (OUT_BF16) ((__bf16*)Cv)[(size_t)row * DIM + col] = (__bf16)acc[i][j][v];
                else          ((float*)Cv)[(size_t)row * DIM + col] = acc[i][j][v];
            }
}

// ---------------------------------------------------------------------------
// Interleaved RoPE in-place on bf16 (NTOK, DIM) viewed as (tok, head, 64 pairs)
// ---------------------------------------------------------------------------
__global__ __launch_bounds__(256) void rope_kernel(void* tv) {
    __bf16* t = (__bf16*)tv;
    size_t idx = (size_t)blockIdx.x * 256 + threadIdx.x;  // NTOK*DIM/2 threads
    int token = (int)(idx >> 11);
    int wi    = (int)(idx & 2047);
    int h     = wi >> 6;
    int d2    = wi & 63;
    int pos   = token & (SEQ - 1);
    float freq = (float)pos * __powf(10000.0f, -(float)(2 * d2) * (1.0f / 128.0f));
    float s, c;
    __sincosf(freq, &s, &c);
    size_t off = (size_t)token * DIM + h * HD + d2 * 2;
    float x0 = (float)t[off], x1 = (float)t[off + 1];
    t[off]     = (__bf16)(x0 * c - x1 * s);
    t[off + 1] = (__bf16)(x0 * s + x1 * c);
}

// ---------------------------------------------------------------------------
// Flash attention: grid (SEQ/128, HEADS, BSZ), 8 waves x 16 Q rows each.
// Double-buffered K/V staging; K async to LDS, V transposed on store.
// ---------------------------------------------------------------------------
__global__ __launch_bounds__(256) void flash_attn(const void* qv, const void* kv_,
                                                  const void* vv, void* ov) {
    const __bf16* Q = (const __bf16*)qv;
    const __bf16* K = (const __bf16*)kv_;
    const __bf16* V = (const __bf16*)vv;
    __bf16*       O = (__bf16*)ov;

    __shared__ __bf16 Klds[2][32 * 136];      // 32 kv rows x 128 d (+8 pad)
    __shared__ __bf16 Vtlds[2][128 * 40];     // transposed: 128 d x 32 kv (+8 pad)
    __shared__ __bf16 Plds[8][16 * 32];       // per-wave probs re-layout buffer

    const int tid  = threadIdx.x;
    const int lane = tid & 31;
    const int wave = tid >> 5;
    const int half = lane >> 4;
    const int cl   = lane & 15;
    const int b  = blockIdx.z;
    const int h  = blockIdx.y;
    const int qb = blockIdx.x;
    const int qrow0 = qb * 128 + wave * 16;

    // Q fragments (A-operand layout): rows = cl, K-chunks per CDNA5 spec
    const __bf16* qptr = Q + ((size_t)(b * SEQ + qrow0 + cl)) * DIM + h * HD;
    v16bf aq[4];
#pragma unroll
    for (int kk = 0; kk < 4; kk++)
        aq[kk] = make_frag(qptr + kk * 32 + half * 8, qptr + kk * 32 + half * 8 + 16);

    v8f acc[8] = {};
    float mrow[8], lrow[8];
#pragma unroll
    for (int i = 0; i < 8; i++) { mrow[i] = -1e30f; lrow[i] = 0.0f; }

    const int krow = tid >> 3;          // 0..31
    const int kcol = (tid & 7) * 16;    // 0..112
    const float sc = 0.08838834764831845f;  // 1/sqrt(128)

    auto stage = [&](int kv0, int buf) {
        const __bf16* kg = K + ((size_t)(b * SEQ + kv0 + krow)) * DIM + h * HD + kcol;
        async_copy32(lds_off(&Klds[buf][krow * 136 + kcol]), kg);
        const __bf16* vg = V + ((size_t)(b * SEQ + kv0 + krow)) * DIM + h * HD + kcol;
        V16U uv; uv.v = *(const v16bf*)vg;
#pragma unroll
        for (int e = 0; e < 16; e++) Vtlds[buf][(kcol + e) * 40 + krow] = uv.e[e];
    };

    const int NB = ((qb + 1) * 128) / 32;
    stage(0, 0);

    for (int it = 0; it < NB; ++it) {
        const int kv0 = it * 32;
        const int buf = it & 1;
        wait_async0();
        __syncthreads();
        if (it + 1 < NB) stage(kv0 + 32, buf ^ 1);

        // Preload all K fragments, then burst the score WMMAs.
        v16bf kb[8];
#pragma unroll
        for (int nt = 0; nt < 2; nt++)
#pragma unroll
            for (int kk = 0; kk < 4; kk++) {
                int r = nt * 16 + cl;
                kb[nt * 4 + kk] = make_frag(
                    &Klds[buf][r * 136 + kk * 32 + half * 8],
                    &Klds[buf][r * 136 + kk * 32 + half * 8 + 16]);
            }
        v8f s0 = {}, s1 = {};
#pragma unroll
        for (int kk = 0; kk < 4; kk++) {
            s0 = __builtin_amdgcn_wmma_f32_16x16x32_bf16(false, aq[kk], false, kb[kk],
                                                         (short)0, s0, false, false);
            s1 = __builtin_amdgcn_wmma_f32_16x16x32_bf16(false, aq[kk], false, kb[4 + kk],
                                                         (short)0, s1, false, false);
        }

        // Online softmax update (row = i + 8*half, col = lane&15 within tile)
#pragma unroll
        for (int i = 0; i < 8; i++) {
            int grow = qrow0 + i + 8 * half;
            float v0 = s0[i] * sc; if (kv0 + cl      > grow) v0 = -1e30f;
            float v1 = s1[i] * sc; if (kv0 + 16 + cl > grow) v1 = -1e30f;
            float mx = fmaxf(v0, v1);
            mx = fmaxf(mx, __shfl_xor(mx, 1, 32));
            mx = fmaxf(mx, __shfl_xor(mx, 2, 32));
            mx = fmaxf(mx, __shfl_xor(mx, 4, 32));
            mx = fmaxf(mx, __shfl_xor(mx, 8, 32));
            float mn = fmaxf(mrow[i], mx);
            float scale = __expf(mrow[i] - mn);
            float p0 = __expf(v0 - mn);
            float p1 = __expf(v1 - mn);
            float rs = p0 + p1;
            rs += __shfl_xor(rs, 1, 32);
            rs += __shfl_xor(rs, 2, 32);
            rs += __shfl_xor(rs, 4, 32);
            rs += __shfl_xor(rs, 8, 32);
            lrow[i] = lrow[i] * scale + rs;
            mrow[i] = mn;
#pragma unroll
            for (int t = 0; t < 8; t++) acc[t][i] *= scale;
            int prow = i + 8 * half;
            Plds[wave][prow * 32 + cl]      = (__bf16)p0;
            Plds[wave][prow * 32 + 16 + cl] = (__bf16)p1;
        }

        // P as A-operand (intra-wave LDS round trip; DS is in-order per wave),
        // V fragments preloaded, then burst the PV WMMAs.
        v16bf ap = make_frag(&Plds[wave][cl * 32 + half * 8],
                             &Plds[wave][cl * 32 + half * 8 + 16]);
        v16bf vb[8];
#pragma unroll
        for (int t = 0; t < 8; t++) {
            int d = t * 16 + cl;
            vb[t] = make_frag(&Vtlds[buf][d * 40 + half * 8],
                              &Vtlds[buf][d * 40 + half * 8 + 16]);
        }
#pragma unroll
        for (int t = 0; t < 8; t++)
            acc[t] = __builtin_amdgcn_wmma_f32_16x16x32_bf16(false, ap, false, vb[t],
                                                             (short)0, acc[t], false, false);
    }

    // Epilogue: normalize and store bf16 attention output (token, dim) layout
#pragma unroll
    for (int i = 0; i < 8; i++) {
        float inv = 1.0f / lrow[i];
        int row = b * SEQ + qrow0 + i + 8 * half;
#pragma unroll
        for (int t = 0; t < 8; t++)
            O[(size_t)row * DIM + h * HD + t * 16 + cl] = (__bf16)(acc[t][i] * inv);
    }
}

// ---------------------------------------------------------------------------
// Host launcher
// ---------------------------------------------------------------------------
extern "C" void kernel_launch(void* const* d_in, const int* in_sizes, int n_in,
                              void* d_out, int out_size, void* d_ws, size_t ws_size,
                              hipStream_t stream) {
    (void)in_sizes; (void)n_in; (void)out_size; (void)ws_size;
    const void* x  = d_in[0];
    // d_in[1] = start_pos (always 0 here)
    const void* wq = d_in[2];
    const void* wk = d_in[3];
    const void* wv = d_in[4];
    const void* wo = d_in[5];

    char* ws = (char*)d_ws;
    const size_t SZ = (size_t)DIM * NTOK * sizeof(__bf16);  // 32 MB
    void* xb   = ws + 0 * SZ;
    void* wqb  = ws + 1 * SZ;
    void* wkb  = ws + 2 * SZ;
    void* wvb  = ws + 3 * SZ;
    void* wob  = ws + 4 * SZ;
    void* qbuf = ws + 5 * SZ;
    void* kbuf = ws + 6 * SZ;
    void* vbuf = ws + 7 * SZ;
    void* abuf = ws + 8 * SZ;

    dim3 blk(256);
    int cvtBlocks = (DIM * NTOK) / (256 * 8);  // 8192
    cvt_f32_bf16<<<cvtBlocks, blk, 0, stream>>>(x,  xb);
    cvt_f32_bf16<<<cvtBlocks, blk, 0, stream>>>(wq, wqb);
    cvt_f32_bf16<<<cvtBlocks, blk, 0, stream>>>(wk, wkb);
    cvt_f32_bf16<<<cvtBlocks, blk, 0, stream>>>(wv, wvb);
    cvt_f32_bf16<<<cvtBlocks, blk, 0, stream>>>(wo, wob);

    dim3 ggrid(DIM / 128, NTOK / 128);
    gemm_nt<true><<<ggrid, blk, 0, stream>>>(xb, wqb, qbuf);
    gemm_nt<true><<<ggrid, blk, 0, stream>>>(xb, wkb, kbuf);
    gemm_nt<true><<<ggrid, blk, 0, stream>>>(xb, wvb, vbuf);

    int ropeBlocks = (NTOK * DIM / 2) / 256;   // 32768
    rope_kernel<<<ropeBlocks, blk, 0, stream>>>(qbuf);
    rope_kernel<<<ropeBlocks, blk, 0, stream>>>(kbuf);

    flash_attn<<<dim3(SEQ / 128, HEADS, BSZ), blk, 0, stream>>>(qbuf, kbuf, vbuf, abuf);

    gemm_nt<false><<<ggrid, blk, 0, stream>>>(abuf, wob, d_out);
}
